// RotaryAttention_67070209294520
// MI455X (gfx1250) — matrix operand order repaired and checked by reading, not compiled, not run on hardware
//
#include <hip/hip_runtime.h>
#include <hip/hip_bf16.h>

// ---------------------------------------------------------------------------
// RotaryAttention for MI455X (gfx1250): bf16 WMMA, LDS-staged double-buffered
// GEMMs using gfx1250 async-to-LDS copies (ASYNCcnt) when available.
// ---------------------------------------------------------------------------

typedef __bf16 bf16;
typedef __attribute__((ext_vector_type(16))) __bf16 v16bf;
typedef __attribute__((ext_vector_type(8)))  __bf16 v8bf;
typedef __attribute__((ext_vector_type(8)))  float  v8f;
typedef int v4i __attribute__((vector_size(16)));   // 16-byte chunk type
typedef __attribute__((address_space(1))) v4i* gv4i_p;   // global
typedef __attribute__((address_space(3))) v4i* lv4i_p;   // LDS

#define B_     8
#define S_     512
#define D_     2048
#define H_     16
#define G_     2
#define HD_    128
#define T_     768
#define CACHE_ 256
#define MM_    (B_ * S_)        // 4096 rows
#define GH_    (G_ * HD_)       // 256
#define SCALE_ 0.08838834764831845f   // 1/sqrt(128)
#define LN1E4_DIV64 0.14391156831f    // ln(10000)/64

#if defined(__has_builtin)
#if __has_builtin(__builtin_amdgcn_global_load_async_to_lds_b128)
#define ASYNC_LDS 1
#endif
#endif

static __device__ __forceinline__ v8f wmma_bf16(v16bf a, v16bf b, v8f c) {
  return __builtin_amdgcn_wmma_f32_16x16x32_bf16(false, a, false, b, (short)0, c,
                                                 false, false);
}

// A-operand fragment (16x32, M row per lane): row = lane&15,
// 16 contiguous K starting at lh*16. p = rowbase+k0.
static __device__ __forceinline__ v16bf load_a16(const bf16* p, int lh) {
  union { v16bf v; v8bf h[2]; } u;
  u.h[0] = *(const v8bf*)(p + lh * 16);
  u.h[1] = *(const v8bf*)(p + lh * 16 + 8);
  return u.v;
}
// B-operand fragment (32x16, N col per lane): col = lane&15,
// 16 contiguous K starting at lh*16 (column-major storage). p = colbase+k0.
static __device__ __forceinline__ v16bf load_b16(const bf16* p, int lh) {
  union { v16bf v; v8bf h[2]; } u;
  const bf16* q = p + lh * 16;
  u.h[0] = *(const v8bf*)(q);
  u.h[1] = *(const v8bf*)(q + 8);
  return u.v;
}

// 16-byte global -> LDS copy; async (ASYNCcnt-tracked) on gfx1250 toolchains
// that expose the builtin, otherwise synchronous reg bounce.
static __device__ __forceinline__ void copy16_g2l(const bf16* g, bf16* l) {
#ifdef ASYNC_LDS
  __builtin_amdgcn_global_load_async_to_lds_b128(
      (gv4i_p)(bf16*)g, (lv4i_p)l, 0, 0);
#else
  *(v8bf*)l = *(const v8bf*)g;
#endif
}
static __device__ __forceinline__ void wait_async_le4() {
#ifdef ASYNC_LDS
#if __has_builtin(__builtin_amdgcn_s_wait_asynccnt)
  __builtin_amdgcn_s_wait_asynccnt(4);
#else
  asm volatile("s_wait_asynccnt 4" ::: "memory");
#endif
#endif
}
static __device__ __forceinline__ void wait_async_0() {
#ifdef ASYNC_LDS
#if __has_builtin(__builtin_amdgcn_s_wait_asynccnt)
  __builtin_amdgcn_s_wait_asynccnt(0);
#else
  asm volatile("s_wait_asynccnt 0" ::: "memory");
#endif
#endif
}

// ------------------------- elementwise cast f32 -> bf16 ---------------------
__global__ __launch_bounds__(256) void cast_bf16_kernel(
    const float4* __restrict__ in, bf16* __restrict__ out, int n4) {
  int i = blockIdx.x * 256 + threadIdx.x;
  if (i >= n4) return;
  float4 f = in[i];
  out[4 * i + 0] = (bf16)f.x;
  out[4 * i + 1] = (bf16)f.y;
  out[4 * i + 2] = (bf16)f.z;
  out[4 * i + 3] = (bf16)f.w;
}

// ----------------- transpose-cast: in[K][N] f32 -> out[N][K] bf16 -----------
__global__ __launch_bounds__(256) void transpose_cast_kernel(
    const float* __restrict__ in, bf16* __restrict__ out, int K, int N) {
  __shared__ float tile[32][33];
  const int bx = blockIdx.x * 32;  // N offset
  const int by = blockIdx.y * 32;  // K offset
  const int tx = threadIdx.x & 31;
  const int ty = threadIdx.x >> 5;  // 0..7
#pragma unroll
  for (int i = ty; i < 32; i += 8)
    tile[i][tx] = in[(size_t)(by + i) * N + bx + tx];
  __syncthreads();
#pragma unroll
  for (int i = ty; i < 32; i += 8)
    out[(size_t)(bx + i) * K + by + tx] = (bf16)tile[tx][i];
}

// --------------------------- WMMA GEMM (bf16 -> f32) ------------------------
// C[M][N] = A[M][K] @ BT[N][K] + bias.  Block tile 128x128, K-step 32,
// double-buffered LDS staging (async-to-LDS), 8 waves as 4(M)x2(N),
// each wave a 32x64 register tile (2x4 WMMA C-tiles).
__global__ __launch_bounds__(256) void gemm_bf16_kernel(
    const bf16* __restrict__ A, const bf16* __restrict__ BT,
    const float* __restrict__ bias, float* __restrict__ C,
    int Ndim, int Kdim) {
  __shared__ __align__(16) bf16 sA[2][128][32];  // 8 KB per buffer
  __shared__ __align__(16) bf16 sB[2][128][32];  // 8 KB per buffer

  const int tid = threadIdx.x;
  const int lane = tid & 31;
  const int wave = tid >> 5;
  const int lr = lane & 15;
  const int lh = lane >> 4;
  const int wm = wave >> 1;  // 0..3 -> M offset wm*32
  const int wn = wave & 1;   // 0..1 -> N offset wn*64
  const int rowblk = blockIdx.y * 128;
  const int colblk = blockIdx.x * 128;

  // Each thread stages 2x16B of A and 2x16B of B per K-step (4 async ops).
  auto stage = [&](int buf, int kb) {
#pragma unroll
    for (int t = 0; t < 2; ++t) {
      int c = tid + t * 256;        // 0..511 chunks of 16B
      int row = c >> 2;             // 0..127
      int ko = (c & 3) * 8;         // element offset in K (0,8,16,24)
      copy16_g2l(&A[(size_t)(rowblk + row) * Kdim + kb + ko], &sA[buf][row][ko]);
      copy16_g2l(&BT[(size_t)(colblk + row) * Kdim + kb + ko], &sB[buf][row][ko]);
    }
  };

  v8f acc[2][4];
#pragma unroll
  for (int m = 0; m < 2; ++m)
#pragma unroll
    for (int n = 0; n < 4; ++n) acc[m][n] = v8f{};

  const int nK = Kdim >> 5;
  stage(0, 0);

  for (int ks = 0; ks < nK; ++ks) {
    const int cur = ks & 1;
    if (ks + 1 < nK) {
      stage(cur ^ 1, (ks + 1) << 5);  // issue next stage before waiting
      wait_async_le4();               // in-order: first 4 (current) are done
    } else {
      wait_async_0();
    }
    __syncthreads();  // current buffer visible block-wide

    v16bf af[2], bfr[4];
#pragma unroll
    for (int m = 0; m < 2; ++m)
      af[m] = load_a16(&sA[cur][wm * 32 + m * 16 + lr][0], lh);
#pragma unroll
    for (int n = 0; n < 4; ++n)
      bfr[n] = load_b16(&sB[cur][wn * 64 + n * 16 + lr][0], lh);
#pragma unroll
    for (int m = 0; m < 2; ++m)
#pragma unroll
      for (int n = 0; n < 4; ++n)
        acc[m][n] = wmma_bf16(af[m], bfr[n], acc[m][n]);

    __syncthreads();  // all reads of `cur` done before it is restaged
  }

#pragma unroll
  for (int m = 0; m < 2; ++m) {
    const int r0 = rowblk + wm * 32 + m * 16 + lh * 8;
#pragma unroll
    for (int n = 0; n < 4; ++n) {
      const int c = colblk + wn * 64 + n * 16 + lr;
      const float bv = bias[c];
#pragma unroll
      for (int j = 0; j < 8; ++j)
        C[(size_t)(r0 + j) * Ndim + c] = acc[m][n][j] + bv;
    }
  }
}

// ------------------- RoPE on Q (+ fold softmax scale), relayout -------------
// q_f32: [(b*S+s)][h*HD+j]  ->  qb bf16: [b][h][s][j]
__global__ __launch_bounds__(256) void rope_q_kernel(
    const float* __restrict__ q, bf16* __restrict__ qb) {
  int idx = blockIdx.x * 256 + threadIdx.x;  // B*S*H*HD threads
  int j = idx & 127;
  int h = (idx >> 7) & 15;
  int s = (idx >> 11) & 511;
  int b = idx >> 20;
  float p = (float)(CACHE_ + s);
  float freq = p * __expf(-LN1E4_DIV64 * (float)(j & 63));
  float sn, cs;
  __sincosf(freq, &sn, &cs);
  const float* src = q + (size_t)(b * S_ + s) * D_ + h * HD_;
  float xj = src[j];
  float xo = (j < 64) ? -src[j + 64] : src[j - 64];
  float val = (xj * cs + xo * sn) * SCALE_;
  qb[(((size_t)(b * H_ + h)) * S_ + s) * HD_ + j] = (bf16)val;
}

// ------- K: RoPE + cache concat; V: concat. fp32 outputs + bf16 copies ------
__global__ __launch_bounds__(256) void kv_concat_kernel(
    const float* __restrict__ knew, const float* __restrict__ vnew,
    const float* __restrict__ pk, const float* __restrict__ pv,
    float* __restrict__ kout, float* __restrict__ vout,
    bf16* __restrict__ kb, bf16* __restrict__ vT) {
  int idx = blockIdx.x * 256 + threadIdx.x;  // B*T*G*HD threads
  int j = idx & 127;
  int g = (idx >> 7) & 1;
  int rest = idx >> 8;  // b*T + t
  int t = rest % T_;
  int b = rest / T_;

  float kval, vval;
  if (t < CACHE_) {
    size_t ci = (((size_t)b * CACHE_ + t) * G_ + g) * HD_ + j;
    kval = pk[ci];
    vval = pv[ci];
  } else {
    int s = t - CACHE_;
    size_t ni = ((size_t)(b * S_ + s)) * GH_ + g * HD_ + j;
    size_t no = ((size_t)(b * S_ + s)) * GH_ + g * HD_ + ((j < 64) ? j + 64 : j - 64);
    float freq = (float)t * __expf(-LN1E4_DIV64 * (float)(j & 63));
    float sn, cs;
    __sincosf(freq, &sn, &cs);
    float xj = knew[ni];
    float xo = (j < 64) ? -knew[no] : knew[no];
    kval = xj * cs + xo * sn;
    vval = vnew[ni];
  }
  size_t oi = (((size_t)b * T_ + t) * G_ + g) * HD_ + j;
  kout[oi] = kval;
  vout[oi] = vval;
  kb[(((size_t)(b * G_ + g)) * T_ + t) * HD_ + j] = (bf16)kval;
  vT[(((size_t)(b * G_ + g)) * HD_ + j) * T_ + t] = (bf16)vval;
}

// --------------------------- flash attention (WMMA) -------------------------
__global__ __launch_bounds__(128) void flash_attn_kernel(
    const bf16* __restrict__ qb, const bf16* __restrict__ kb,
    const bf16* __restrict__ vT, bf16* __restrict__ ctx) {
  __shared__ __align__(16) bf16 pshm_all[4][16 * 32];
  const int lane = threadIdx.x & 31;
  const int wave = threadIdx.x >> 5;
  bf16* pshm = pshm_all[wave];
  const int lr = lane & 15;
  const int lh = lane >> 4;

  int wid = blockIdx.x * 4 + wave;  // 0 .. B*H*32-1
  int qt = wid & 31;
  int h = (wid >> 5) & 15;
  int b = wid >> 9;
  int g = h >> 3;  // h / (H/G)

  const bf16* qbase = qb + (((size_t)(b * H_ + h)) * S_ + qt * 16 + lr) * HD_;
  v16bf qf[4];
#pragma unroll
  for (int kc = 0; kc < 4; ++kc) qf[kc] = load_a16(qbase + kc * 32, lh);

  v8f O[8] = {v8f{}, v8f{}, v8f{}, v8f{}, v8f{}, v8f{}, v8f{}, v8f{}};
  float m[8], l[8], alpha[8];
#pragma unroll
  for (int j = 0; j < 8; ++j) { m[j] = -3.0e38f; l[j] = 0.f; }

  const bf16* kbase = kb + ((size_t)(b * G_ + g)) * T_ * HD_;
  const bf16* vbase = vT + ((size_t)(b * G_ + g)) * HD_ * T_;

  for (int t0 = 0; t0 < T_; t0 += 32) {
    v8f s0 = v8f{}, s1 = v8f{};
#pragma unroll
    for (int kc = 0; kc < 4; ++kc) {
      const bf16* k0 = kbase + (size_t)(t0 + lr) * HD_ + kc * 32;
      const bf16* k1 = kbase + (size_t)(t0 + 16 + lr) * HD_ + kc * 32;
      s0 = wmma_bf16(qf[kc], load_b16(k0, lh), s0);
      s1 = wmma_bf16(qf[kc], load_b16(k1, lh), s1);
    }
#pragma unroll
    for (int j = 0; j < 8; ++j) {
      float a0 = s0[j], a1 = s1[j];
      float mx = fmaxf(a0, a1);
#pragma unroll
      for (int o = 8; o > 0; o >>= 1) mx = fmaxf(mx, __shfl_xor(mx, o, 32));
      float mn = fmaxf(m[j], mx);
      float al = __expf(m[j] - mn);
      float p0 = __expf(a0 - mn);
      float p1 = __expf(a1 - mn);
      float rs = p0 + p1;
#pragma unroll
      for (int o = 8; o > 0; o >>= 1) rs += __shfl_xor(rs, o, 32);
      l[j] = l[j] * al + rs;
      m[j] = mn;
      alpha[j] = al;
      pshm[(lh * 8 + j) * 32 + lr] = (bf16)p0;
      pshm[(lh * 8 + j) * 32 + 16 + lr] = (bf16)p1;
    }
#pragma unroll
    for (int n = 0; n < 8; ++n)
#pragma unroll
      for (int j = 0; j < 8; ++j) O[n][j] *= alpha[j];

    asm volatile("s_wait_dscnt 0" ::: "memory");

    v16bf pf = load_a16(pshm + lr * 32, lh);
#pragma unroll
    for (int n = 0; n < 8; ++n) {
      const bf16* v0 = vbase + (size_t)(n * 16 + lr) * T_ + t0;
      O[n] = wmma_bf16(pf, load_b16(v0, lh), O[n]);
    }
  }

  float invl[8];
#pragma unroll
  for (int j = 0; j < 8; ++j) invl[j] = 1.0f / l[j];

  bf16* cb = ctx + ((size_t)(b * S_) + qt * 16) * D_ + h * HD_;
#pragma unroll
  for (int n = 0; n < 8; ++n)
#pragma unroll
    for (int j = 0; j < 8; ++j)
      cb[(size_t)(lh * 8 + j) * D_ + n * 16 + lr] = (bf16)(O[n][j] * invl[j]);
}

// ---------------------------------------------------------------------------
extern "C" void kernel_launch(void* const* d_in, const int* in_sizes, int n_in,
                              void* d_out, int out_size, void* d_ws, size_t ws_size,
                              hipStream_t stream) {
  const float* x  = (const float*)d_in[0];
  const float* pk = (const float*)d_in[1];
  const float* pv = (const float*)d_in[2];
  const float* Wq = (const float*)d_in[3];
  const float* bq = (const float*)d_in[4];
  const float* Wk = (const float*)d_in[5];
  const float* bk = (const float*)d_in[6];
  const float* Wv = (const float*)d_in[7];
  const float* bv = (const float*)d_in[8];
  const float* Wo = (const float*)d_in[9];
  const float* bo = (const float*)d_in[10];

  float* out  = (float*)d_out;
  float* kout = out + (size_t)B_ * S_ * D_;
  float* vout = kout + (size_t)B_ * T_ * G_ * HD_;

  char* ws = (char*)d_ws;
  size_t off = 0;
  auto alloc = [&](size_t bytes) -> void* {
    void* p = ws + off;
    off += (bytes + 255) & ~(size_t)255;
    return p;
  };
  bf16*  xb    = (bf16*)alloc((size_t)MM_ * D_ * 2);
  bf16*  WqT   = (bf16*)alloc((size_t)D_ * D_ * 2);
  bf16*  WkT   = (bf16*)alloc((size_t)GH_ * D_ * 2);
  bf16*  WvT   = (bf16*)alloc((size_t)GH_ * D_ * 2);
  bf16*  WoT   = (bf16*)alloc((size_t)D_ * D_ * 2);
  float* q_f32 = (float*)alloc((size_t)MM_ * D_ * 4);
  float* k_new = (float*)alloc((size_t)MM_ * GH_ * 4);
  float* v_new = (float*)alloc((size_t)MM_ * GH_ * 4);
  bf16*  qbuf  = (bf16*)alloc((size_t)MM_ * D_ * 2);
  bf16*  kbuf  = (bf16*)alloc((size_t)B_ * G_ * T_ * HD_ * 2);
  bf16*  vTb   = (bf16*)alloc((size_t)B_ * G_ * T_ * HD_ * 2);
  bf16*  ctx   = (bf16*)q_f32;  // alias: q_f32 dead after rope_q

  cast_bf16_kernel<<<(MM_ * D_ / 4 + 255) / 256, 256, 0, stream>>>(
      (const float4*)x, xb, MM_ * D_ / 4);
  transpose_cast_kernel<<<dim3(D_ / 32, D_ / 32), 256, 0, stream>>>(Wq, WqT, D_, D_);
  transpose_cast_kernel<<<dim3(GH_ / 32, D_ / 32), 256, 0, stream>>>(Wk, WkT, D_, GH_);
  transpose_cast_kernel<<<dim3(GH_ / 32, D_ / 32), 256, 0, stream>>>(Wv, WvT, D_, GH_);
  transpose_cast_kernel<<<dim3(D_ / 32, D_ / 32), 256, 0, stream>>>(Wo, WoT, D_, D_);

  gemm_bf16_kernel<<<dim3(D_ / 128, MM_ / 128), 256, 0, stream>>>(
      xb, WqT, bq, q_f32, D_, D_);
  gemm_bf16_kernel<<<dim3(GH_ / 128, MM_ / 128), 256, 0, stream>>>(
      xb, WkT, bk, k_new, GH_, D_);
  gemm_bf16_kernel<<<dim3(GH_ / 128, MM_ / 128), 256, 0, stream>>>(
      xb, WvT, bv, v_new, GH_, D_);

  rope_q_kernel<<<(B_ * S_ * H_ * HD_) / 256, 256, 0, stream>>>(q_f32, qbuf);
  kv_concat_kernel<<<(B_ * T_ * G_ * HD_) / 256, 256, 0, stream>>>(
      k_new, v_new, pk, pv, kout, vout, kbuf, vTb);

  flash_attn_kernel<<<(B_ * H_ * (S_ / 16)) / 4, 128, 0, stream>>>(
      qbuf, kbuf, vTb, ctx);

  gemm_bf16_kernel<<<dim3(D_ / 128, MM_ / 128), 256, 0, stream>>>(
      ctx, WoT, bo, out, D_, D_);
}
